// Head_39256001085546
// MI455X (gfx1250) — compile-verified, matrix operand order checked by reading, not run on hardware
//
#include <hip/hip_runtime.h>
#include <hip/hip_bf16.h>
#include <stdint.h>

typedef __attribute__((ext_vector_type(16))) _Float16 v16h;
typedef __attribute__((ext_vector_type(8)))  _Float16 v8h;
typedef __attribute__((ext_vector_type(8)))  float    v8f;
typedef __attribute__((ext_vector_type(4)))  unsigned v4u;
typedef __attribute__((ext_vector_type(8)))  unsigned v8u;

#define B_   4
#define T_   4096
#define C_   1024
#define HS_  64
#define NELEM (B_ * T_ * HS_)

__device__ __forceinline__ v8f wmma_f16(v16h a, v16h b, v8f c) {
  // D = A(16x32 f16) * B(32x16 f16) + C(16x16 f32)
  return __builtin_amdgcn_wmma_f32_16x16x32_f16(
      false, a, false, b, (short)0, c, false, false);
}

// Load 16 fp32 values in the 16-bit A/B per-lane K pattern
// (K = kbl+{0..7} and kbl+16+{0..7}), convert to f16.
__device__ __forceinline__ v16h cvt16(const float* p, int kbl) {
  float4 f0 = *(const float4*)(p + kbl);
  float4 f1 = *(const float4*)(p + kbl + 4);
  float4 f2 = *(const float4*)(p + kbl + 16);
  float4 f3 = *(const float4*)(p + kbl + 20);
  v16h r = {};
  r[0]  = (_Float16)f0.x; r[1]  = (_Float16)f0.y; r[2]  = (_Float16)f0.z; r[3]  = (_Float16)f0.w;
  r[4]  = (_Float16)f1.x; r[5]  = (_Float16)f1.y; r[6]  = (_Float16)f1.z; r[7]  = (_Float16)f1.w;
  r[8]  = (_Float16)f2.x; r[9]  = (_Float16)f2.y; r[10] = (_Float16)f2.z; r[11] = (_Float16)f2.w;
  r[12] = (_Float16)f3.x; r[13] = (_Float16)f3.y; r[14] = (_Float16)f3.z; r[15] = (_Float16)f3.w;
  return r;
}

// Same pattern but from f16 LDS data (two contiguous 16B segments).
__device__ __forceinline__ v16h pack16(const _Float16* p, int kbl) {
  v8h lo = *(const v8h*)(p + kbl);
  v8h hi = *(const v8h*)(p + kbl + 16);
  v16h r = {};
#pragma unroll
  for (int i = 0; i < 8; ++i) { r[i] = lo[i]; r[i + 8] = hi[i]; }
  return r;
}

// Coalesced write-back of a 16x64 fp32 tile staged in LDS:
// lane -> row = lane>>1, cols = (lane&1)*32 .. +31, 8x async b128 stores.
__device__ __forceinline__ void async_store_tile(const float* os, float* gdst,
                                                 int lane) {
  const unsigned lbase =
      (unsigned)(uintptr_t)os + (unsigned)(((lane >> 1) * HS_ + (lane & 1) * 32) * 4);
  float* gp = gdst + (size_t)(lane >> 1) * HS_ + (lane & 1) * 32;
#pragma unroll
  for (int j = 0; j < 8; ++j) {
    uint64_t ga = (uint64_t)(uintptr_t)(gp + j * 4);
    unsigned la = lbase + (unsigned)(j * 16);
    asm volatile("global_store_async_from_lds_b128 %0, %1, off"
                 :: "v"(ga), "v"(la) : "memory");
  }
  asm volatile("s_wait_asynccnt 0x0" ::: "memory");
}

// ---------------------------------------------------------------------------
// Kernel 1: q/k/v = x @ W{q,k,v}.  grid=(128,3), block=256 (8 waves).
// Each wave: one 16-row tile, all 4 HS n-tiles. W K-slice staged in LDS (f16,
// transposed) and shared by all 8 waves.  q -> d_out[0:N] (scratch region
// later overwritten in place by attention out), k -> [N:2N], v -> [2N:3N].
// ---------------------------------------------------------------------------
__global__ __launch_bounds__(256) void qkv_proj_kernel(
    const float* __restrict__ x, const float* __restrict__ Wq,
    const float* __restrict__ Wk, const float* __restrict__ Wv,
    float* __restrict__ out) {
  __shared__ _Float16 Wt[HS_][32];      // [n][c-c0], f16
  __shared__ float    oS1[8][16 * HS_]; // per-wave output staging

  const int tid  = threadIdx.x;
  const int lane = tid & 31;
  const int wave = tid >> 5;
  const int m    = lane & 15;
  const int half = lane >> 4;
  const int kbl  = half ? 8 : 0;

  const int mat = blockIdx.y;
  const float* W = (mat == 0) ? Wq : (mat == 1) ? Wk : Wv;
  float* dst = out + (size_t)mat * NELEM;

  const int row0 = (blockIdx.x * 8 + wave) * 16;
  const int cw = tid >> 3;        // 0..31 : K row within slice
  const int n0 = (tid & 7) * 8;   // 0..56 : N column group

  v8f acc[4] = {};

  for (int kk = 0; kk < C_ / 32; ++kk) {
    const int c0 = kk * 32;
    __syncthreads();
    {  // stage W[c0..c0+31][0..63] transposed to f16
      const float* src = W + (size_t)(c0 + cw) * HS_ + n0;
      float4 f0 = *(const float4*)(src);
      float4 f1 = *(const float4*)(src + 4);
      float t8[8] = {f0.x, f0.y, f0.z, f0.w, f1.x, f1.y, f1.z, f1.w};
#pragma unroll
      for (int j = 0; j < 8; ++j) Wt[n0 + j][cw] = (_Float16)t8[j];
    }
    __syncthreads();

    v16h A = cvt16(x + (size_t)(row0 + m) * C_ + c0, kbl);
    v16h Bv[4];
#pragma unroll
    for (int nt = 0; nt < 4; ++nt) Bv[nt] = pack16(&Wt[nt * 16 + m][0], kbl);
#pragma unroll
    for (int nt = 0; nt < 4; ++nt) acc[nt] = wmma_f16(A, Bv[nt], acc[nt]);
  }

  // stage C-layout scatter into LDS, then coalesced async write-back
  float* os = &oS1[wave][0];
#pragma unroll
  for (int nt = 0; nt < 4; ++nt)
#pragma unroll
    for (int r = 0; r < 8; ++r)
      os[(r + half * 8) * HS_ + nt * 16 + m] = acc[nt][r];
  asm volatile("s_wait_dscnt 0x0" ::: "memory");
  async_store_tile(os, dst + (size_t)row0 * HS_, lane);
}

// ---------------------------------------------------------------------------
// Kernel 2: causal flash attention.  grid=(T/64, B), block=128 (4 waves);
// each wave owns a 16-query tile, block streams 32-key tiles through LDS.
// K tile fetched by the Tensor Data Mover; V converted/transposed to f16.
// ---------------------------------------------------------------------------
__global__ __launch_bounds__(128) void attn_kernel(
    const float* qg, const float* __restrict__ kg,
    const float* __restrict__ vg, float* og) {
  __shared__ float    kT[32 * HS_];     // fp32 [key][h]  (TDM destination)
  __shared__ _Float16 vT[HS_][32];      // f16  [h][key]
  __shared__ _Float16 pS[4][16 * 32];   // per-wave P staging
  __shared__ float    oS[4][16 * HS_];  // per-wave output staging

  const int tid  = threadIdx.x;
  const int lane = tid & 31;
  const int wave = tid >> 5;
  const int m    = lane & 15;
  const int half = lane >> 4;
  const int kbl  = half ? 8 : 0;

  const int b    = blockIdx.y;
  const int q0   = blockIdx.x * 64;
  const int row0 = q0 + wave * 16;
  const size_t bT = (size_t)b * T_;

  v16h aq[2];
  {
    const float* qr = qg + (bT + row0 + m) * HS_;
    aq[0] = cvt16(qr, kbl);
    aq[1] = cvt16(qr + 32, kbl);
  }

  float mrun[8], lrun[8];
#pragma unroll
  for (int r = 0; r < 8; ++r) { mrun[r] = -3.0e38f; lrun[r] = 0.0f; }
  v8f o[4] = {};

  const int kbmax = (q0 + 63) >> 5;

  for (int kb = 0; kb <= kbmax; ++kb) {
    __syncthreads();
    if (wave == 0) {
      // TDM: DMA the 32x64 fp32 K tile into LDS (2D descriptor, 4B elements)
      uint64_t ga = (uint64_t)(uintptr_t)(kg + (bT + kb * 32) * HS_);
      unsigned lds = (unsigned)(uintptr_t)&kT[0];
      v4u g0;
      g0[0] = 1u;                                              // count=1, load desc
      g0[1] = lds;                                             // lds_addr
      g0[2] = (unsigned)(ga & 0xFFFFFFFFu);                    // global_addr[31:0]
      g0[3] = (unsigned)((ga >> 32) & 0x01FFFFFFu) | (2u << 30);  // addr[56:32] | type=2
      v8u g1;
      g1[0] = (2u << 16);    // data_size = 4B
      g1[1] = (64u << 16);   // tensor_dim0 = 64 (bits 63:48)
      g1[2] = (32u << 16);   // tensor_dim1 = 32 (bits 95:80)
      g1[3] = (64u << 16);   // tile_dim0 = 64 (bits 127:112)
      g1[4] = 32u;           // tile_dim1 = 32
      g1[5] = 64u;           // tensor_dim0_stride = 64
      g1[6] = 0u;
      g1[7] = 0u;
      v4u gz = 0u;
      asm volatile("tensor_load_to_lds %0, %1, %2, %3"
                   :: "s"(g0), "s"(g1), "s"(gz), "s"(gz) : "memory");
      __builtin_amdgcn_s_wait_tensorcnt(0);
    }
    {  // stage V tile transposed as f16
      const int key = tid & 31;
      const int h0  = (tid >> 5) * 16;
      const float* vsrc = vg + (bT + kb * 32 + key) * HS_ + h0;
      float4 g0 = *(const float4*)(vsrc);
      float4 g1 = *(const float4*)(vsrc + 4);
      float4 g2 = *(const float4*)(vsrc + 8);
      float4 g3 = *(const float4*)(vsrc + 12);
      float t16[16] = {g0.x, g0.y, g0.z, g0.w, g1.x, g1.y, g1.z, g1.w,
                       g2.x, g2.y, g2.z, g2.w, g3.x, g3.y, g3.z, g3.w};
#pragma unroll
      for (int j = 0; j < 16; ++j) vT[h0 + j][key] = (_Float16)t16[j];
    }
    __syncthreads();

    if (kb * 32 <= row0 + 15) {  // wave-uniform causal guard
      // S = q * k^T : batch all 4 B operands, then 4 WMMAs back-to-back
      v16h Bk[4];
      const float* kp0 = &kT[m * HS_];
      const float* kp1 = &kT[(16 + m) * HS_];
      Bk[0] = cvt16(kp0, kbl);
      Bk[1] = cvt16(kp0 + 32, kbl);
      Bk[2] = cvt16(kp1, kbl);
      Bk[3] = cvt16(kp1 + 32, kbl);
      v8f sc[2] = {};
      sc[0] = wmma_f16(aq[0], Bk[0], sc[0]);
      sc[0] = wmma_f16(aq[1], Bk[1], sc[0]);
      sc[1] = wmma_f16(aq[0], Bk[2], sc[1]);
      sc[1] = wmma_f16(aq[1], Bk[3], sc[1]);

      // online softmax (row = r + 8*half per C-layout; reduce over 16 lanes)
#pragma unroll
      for (int r = 0; r < 8; ++r) {
        const int row  = row0 + r + half * 8;
        const int col0 = kb * 32 + m;
        float s0 = sc[0][r] * 0.125f;   // 1/sqrt(64)
        float s1 = sc[1][r] * 0.125f;
        if (col0 > row)      s0 = -3.0e38f;
        if (col0 + 16 > row) s1 = -3.0e38f;
        float mx = fmaxf(s0, s1);
#pragma unroll
        for (int off = 1; off < 16; off <<= 1)
          mx = fmaxf(mx, __shfl_xor(mx, off, 16));
        const float mnew  = fmaxf(mrun[r], mx);
        const float alpha = __expf(mrun[r] - mnew);
        const float p0 = __expf(s0 - mnew);
        const float p1 = __expf(s1 - mnew);
        float ps = p0 + p1;
#pragma unroll
        for (int off = 1; off < 16; off <<= 1)
          ps += __shfl_xor(ps, off, 16);
        lrun[r] = lrun[r] * alpha + ps;
        mrun[r] = mnew;
#pragma unroll
        for (int t = 0; t < 4; ++t) o[t][r] *= alpha;
        const int rl = r + half * 8;
        pS[wave][rl * 32 + m]      = (_Float16)p0;
        pS[wave][rl * 32 + 16 + m] = (_Float16)p1;
      }
      asm volatile("s_wait_dscnt 0x0" ::: "memory");  // wave-local C->A relayout
      v16h Pa = pack16(&pS[wave][m * 32], kbl);
      v16h Bv[4];
#pragma unroll
      for (int t = 0; t < 4; ++t) Bv[t] = pack16(&vT[t * 16 + m][0], kbl);
#pragma unroll
      for (int t = 0; t < 4; ++t) o[t] = wmma_f16(Pa, Bv[t], o[t]);
    }
  }

  // epilogue: normalize, stage in LDS, coalesced async write-back
  float* os = &oS[wave][0];
#pragma unroll
  for (int r = 0; r < 8; ++r) {
    const float inv = 1.0f / lrun[r];
#pragma unroll
    for (int t = 0; t < 4; ++t)
      os[(r + half * 8) * HS_ + t * 16 + m] = o[t][r] * inv;
  }
  asm volatile("s_wait_dscnt 0x0" ::: "memory");
  async_store_tile(os, og + (bT + row0) * HS_, lane);
}

extern "C" void kernel_launch(void* const* d_in, const int* in_sizes, int n_in,
                              void* d_out, int out_size, void* d_ws, size_t ws_size,
                              hipStream_t stream) {
  (void)in_sizes; (void)n_in; (void)out_size; (void)d_ws; (void)ws_size;
  const float* x  = (const float*)d_in[0];
  const float* Wk = (const float*)d_in[1];
  const float* Wq = (const float*)d_in[2];
  const float* Wv = (const float*)d_in[3];
  float* out = (float*)d_out;

  // q -> out[0:N] (scratch, overwritten in place), k -> out[N:2N], v -> out[2N:3N]
  qkv_proj_kernel<<<dim3(128, 3), 256, 0, stream>>>(x, Wq, Wk, Wv, out);
  attn_kernel<<<dim3(T_ / 64, B_), 128, 0, stream>>>(
      out, out + NELEM, out + 2 * (size_t)NELEM, out);
}
